// HyenaOperator_45311904973318
// MI455X (gfx1250) — matrix operand order repaired and checked by reading, not compiled, not run on hardware
//
#include <hip/hip_runtime.h>
#include <hip/hip_bf16.h>

typedef __bf16 bf16;
typedef __attribute__((ext_vector_type(16))) __bf16 v16bf;
typedef __attribute__((ext_vector_type(8)))  float  v8f;

#define L_SEQ 4096
#define BATCH 2
#define DIM   2048
#define E3    (3*DIM)
#define N2    8192
#define LOG2N 13
#define PI_F  3.14159265358979323846f

// ---- CDNA5 async global->LDS staging (guarded: falls back to sync path) ----
#if defined(__HIP_DEVICE_COMPILE__) && __has_builtin(__builtin_amdgcn_global_load_async_to_lds_b128)
  #define HAVE_ASYNC_LDS 1
  // builtin signature: (V4i addrspace(1)*, V4i addrspace(3)*, imm offset, imm cpol)
  typedef int v4i_vs __attribute__((vector_size(16)));
  #define GPTR(p) ((__attribute__((address_space(1))) v4i_vs*)(p))
  #define LPTR(p) ((__attribute__((address_space(3))) v4i_vs*)(p))
__device__ __forceinline__ void async_wait0() {
#if __has_builtin(__builtin_amdgcn_s_wait_asynccnt)
  __builtin_amdgcn_s_wait_asynccnt(0);
#else
  asm volatile("s_wait_asynccnt 0" ::: "memory");
#endif
}
#endif

// ---------------- f32 -> bf16 convert ----------------
__global__ __launch_bounds__(256) void cvt_bf16_kernel(const float* __restrict__ in,
                                                       bf16* __restrict__ out, int n) {
  int i = blockIdx.x * 256 + threadIdx.x;
  int stride = gridDim.x * 256;
  for (; i < n; i += stride) out[i] = (bf16)in[i];
}

// ---------------- LayerNorm: u[L][B][D] -> xn bf16 same layout ----------------
__global__ __launch_bounds__(256) void ln_kernel(const float* __restrict__ u,
                                                 const float* __restrict__ g,
                                                 const float* __restrict__ be,
                                                 bf16* __restrict__ xn) {
  int row = blockIdx.x;                       // l*B + b
  const float* x = u + (size_t)row * DIM;
  bf16* y = xn + (size_t)row * DIM;
  int tid = threadIdx.x;
  float v[8];
  float s = 0.f;
  for (int j = 0; j < 8; ++j) { v[j] = x[tid + j * 256]; s += v[j]; }
  __shared__ float red[256];
  red[tid] = s; __syncthreads();
  for (int off = 128; off > 0; off >>= 1) {
    if (tid < off) red[tid] += red[tid + off];
    __syncthreads();
  }
  float mu = red[0] * (1.0f / DIM);
  __syncthreads();
  float s2 = 0.f;
  for (int j = 0; j < 8; ++j) { float d = v[j] - mu; s2 += d * d; }
  red[tid] = s2; __syncthreads();
  for (int off = 128; off > 0; off >>= 1) {
    if (tid < off) red[tid] += red[tid + off];
    __syncthreads();
  }
  float rs = rsqrtf(red[0] * (1.0f / DIM) + 1e-5f);
  for (int j = 0; j < 8; ++j) {
    int d = tid + j * 256;
    y[d] = (bf16)((v[j] - mu) * rs * g[d] + be[d]);
  }
}

// ---------------- GEMM1: up_t[b][e][l] = sum_d xn[l,b,d]*Win[e,d] + b_in[e] ----------------
// Block tile 64(M=l) x 128(N=e), K-chunks of 32, 8 waves, 2x2 16x16 WMMA subtiles/wave.
__global__ __launch_bounds__(256) void gemm1_kernel(const bf16* __restrict__ A,   // xn [L][B][D]
                                                    const bf16* __restrict__ Bw,  // win [3D][D]
                                                    const float* __restrict__ bias,
                                                    float* __restrict__ outT) {   // [B][3D][L]
  const int b = blockIdx.z;
  const int mBase = blockIdx.x * 64;
  const int nBase = blockIdx.y * 128;
  const int lda = BATCH * DIM;
  const bf16* Ab = A + (size_t)b * DIM;
  float* outb = outT + (size_t)b * E3 * L_SEQ;

  __shared__ bf16 As[64 * 40];    // padded rows: 80B stride keeps 16B-aligned chunks
  __shared__ bf16 Bs[128 * 40];

  const int tid = threadIdx.x;
  const int lane = tid & 31;
  const int wave = tid >> 5;
  const int wm = (wave >> 2) * 32;   // 0,32
  const int wn = (wave & 3) * 32;    // 0..96
  const int ln = lane & 15;
  const int hi = lane >> 4;

  v8f acc[2][2] = {};

  const int ar = tid >> 2;           // 0..63
  const int ac = (tid & 3) * 8;      // 0,8,16,24

  for (int kt = 0; kt < DIM; kt += 32) {
#if defined(HAVE_ASYNC_LDS)
    // CDNA5 path: DMA tiles straight into LDS (ASYNCcnt), no VGPR round-trip.
    __builtin_amdgcn_global_load_async_to_lds_b128(
        GPTR(Ab + (size_t)(mBase + ar) * lda + kt + ac), LPTR(&As[ar * 40 + ac]), 0, 0);
    for (int i = 0; i < 2; ++i) {
      int ch = tid + i * 256;
      int brr = ch >> 2, bcc = (ch & 3) * 8;
      __builtin_amdgcn_global_load_async_to_lds_b128(
          GPTR(Bw + (size_t)(nBase + brr) * DIM + kt + bcc), LPTR(&Bs[brr * 40 + bcc]), 0, 0);
    }
    async_wait0();
#else
    uint4 av = *(const uint4*)(Ab + (size_t)(mBase + ar) * lda + kt + ac);
    *(uint4*)&As[ar * 40 + ac] = av;
    for (int i = 0; i < 2; ++i) {
      int ch = tid + i * 256;
      int brr = ch >> 2, bcc = (ch & 3) * 8;
      uint4 bv = *(const uint4*)(Bw + (size_t)(nBase + brr) * DIM + kt + bcc);
      *(uint4*)&Bs[brr * 40 + bcc] = bv;
    }
#endif
    if (kt + 32 < DIM) {
      __builtin_prefetch(Ab + (size_t)(mBase + ar) * lda + kt + 32 + ac, 0, 0);
      __builtin_prefetch(Bw + (size_t)(nBase + (tid >> 2)) * DIM + kt + 32, 0, 0);
    }
    __syncthreads();

    union FR { uint4 q[2]; v16bf v; } fa[2], fb[2];
    const int k0a = hi * 8;   // A: lanes 0-15 -> K 0..7 & 16..23 ; lanes 16-31 -> 8..15 & 24..31
    for (int s = 0; s < 2; ++s) {
      int row = wm + s * 16 + ln;
      fa[s].q[0] = *(const uint4*)&As[row * 40 + k0a];
      fa[s].q[1] = *(const uint4*)&As[row * 40 + k0a + 16];
    }
    const int k0b = hi * 8;   // B: lanes 0-15 -> K 0..7 & 16..23 ; lanes 16-31 -> 8..15 & 24..31
    for (int s = 0; s < 2; ++s) {
      int row = wn + s * 16 + ln;
      fb[s].q[0] = *(const uint4*)&Bs[row * 40 + k0b];
      fb[s].q[1] = *(const uint4*)&Bs[row * 40 + k0b + 16];
    }
    for (int si = 0; si < 2; ++si)
      for (int sj = 0; sj < 2; ++sj)
        acc[si][sj] = __builtin_amdgcn_wmma_f32_16x16x32_bf16(
            false, fa[si].v, false, fb[sj].v, (short)0, acc[si][sj], false, false);
    __syncthreads();
  }

  // epilogue: transposed store up_t[n][m], + bias
  for (int si = 0; si < 2; ++si)
    for (int sj = 0; sj < 2; ++sj) {
      int n = nBase + wn + sj * 16 + ln;
      int m = mBase + wm + si * 16 + hi * 8;
      float bia = bias[n];
      v8f c = acc[si][sj];
      float* p = outb + (size_t)n * L_SEQ + m;
      float4 lo = {c[0] + bia, c[1] + bia, c[2] + bia, c[3] + bia};
      float4 hh = {c[4] + bia, c[5] + bia, c[6] + bia, c[7] + bia};
      *(float4*)p = lo;
      *(float4*)(p + 4) = hh;
    }
}

// ---------------- FFT helpers: DIF forward (natural->bitrev), DIT inverse (bitrev->natural) ----------------
__device__ __forceinline__ void fft_dif_fwd(float* re, float* im, int tid) {
  for (int s = LOG2N; s >= 1; --s) {
    int half = 1 << (s - 1);
    for (int t = tid; t < N2 / 2; t += 256) {
      int j = t & (half - 1);
      int i0 = ((t >> (s - 1)) << s) + j;
      int i1 = i0 + half;
      float ang = -PI_F * (float)j / (float)half;
      float sn, cs;
      __sincosf(ang, &sn, &cs);
      float ar = re[i0], ai = im[i0];
      float br = re[i1], bi = im[i1];
      re[i0] = ar + br; im[i0] = ai + bi;
      float dr = ar - br, di = ai - bi;
      re[i1] = dr * cs - di * sn;
      im[i1] = dr * sn + di * cs;
    }
    __syncthreads();
  }
}

__device__ __forceinline__ void fft_dit_inv(float* re, float* im, int tid) {
  for (int s = 1; s <= LOG2N; ++s) {
    int half = 1 << (s - 1);
    for (int t = tid; t < N2 / 2; t += 256) {
      int j = t & (half - 1);
      int i0 = ((t >> (s - 1)) << s) + j;
      int i1 = i0 + half;
      float ang = PI_F * (float)j / (float)half;
      float sn, cs;
      __sincosf(ang, &sn, &cs);
      float br = re[i1], bi = im[i1];
      float tr = br * cs - bi * sn;
      float ti = br * sn + bi * cs;
      re[i1] = re[i0] - tr; im[i1] = im[i0] - ti;
      re[i0] += tr;         im[i0] += ti;
    }
    __syncthreads();
  }
}

// ---------------- K spectrum (stored bit-reversed order, matching DIF output) ----------------
__global__ __launch_bounds__(256) void fftk_kernel(const float* __restrict__ k,
                                                   float2* __restrict__ kf) {
  __shared__ float re[N2];
  __shared__ float im[N2];
  int d = blockIdx.x, tid = threadIdx.x;
  const float* kr = k + (size_t)d * L_SEQ;
  for (int i = tid; i < N2; i += 256) { re[i] = (i < L_SEQ) ? kr[i] : 0.f; im[i] = 0.f; }
  __syncthreads();
  fft_dif_fwd(re, im, tid);
  float2* o = kf + (size_t)d * N2;
  for (int i = tid; i < N2; i += 256) o[i] = make_float2(re[i], im[i]);
}

// ---------------- fused: short conv + gate v*x1 + FFT long conv + bias passthrough + x0 gate ----------------
__global__ __launch_bounds__(256) void channel_kernel(const float* __restrict__ upT, // [B][3D][L]
                                                      const float* __restrict__ sw,  // [3D][3]
                                                      const float* __restrict__ sb,  // [3D]
                                                      const float2* __restrict__ kf, // [D][N2] bitrev
                                                      const float* __restrict__ conv_bias,
                                                      bf16* __restrict__ yg) {       // [B][D][L]
  __shared__ float re[N2];
  __shared__ float im[N2];
  const int d = blockIdx.x, b = blockIdx.y, tid = threadIdx.x;
  const float* base = upT + (size_t)b * E3 * L_SEQ;
  const float* r0 = base + (size_t)d * L_SEQ;               // x0 channel
  const float* r1 = base + (size_t)(DIM + d) * L_SEQ;       // x1 channel
  const float* r2 = base + (size_t)(2 * DIM + d) * L_SEQ;   // v channel
  const int e0 = d, e1 = DIM + d, e2 = 2 * DIM + d;
  float w00 = sw[e0 * 3 + 0], w01 = sw[e0 * 3 + 1], w02 = sw[e0 * 3 + 2], b0 = sb[e0];
  float w10 = sw[e1 * 3 + 0], w11 = sw[e1 * 3 + 1], w12 = sw[e1 * 3 + 2], b1 = sb[e1];
  float w20 = sw[e2 * 3 + 0], w21 = sw[e2 * 3 + 1], w22 = sw[e2 * 3 + 2], b2 = sb[e2];
  float cb = conv_bias[d];

  float vloc[16], x0loc[16];
  for (int j = 0; j < 16; ++j) {
    int l = j * 256 + tid;
    float m2 = (l >= 2) ? 1.f : 0.f, m1 = (l >= 1) ? 1.f : 0.f;
    float a0m2 = m2 * r0[(l >= 2) ? l - 2 : 0], a0m1 = m1 * r0[(l >= 1) ? l - 1 : 0], a00 = r0[l];
    float a1m2 = m2 * r1[(l >= 2) ? l - 2 : 0], a1m1 = m1 * r1[(l >= 1) ? l - 1 : 0], a10 = r1[l];
    float a2m2 = m2 * r2[(l >= 2) ? l - 2 : 0], a2m1 = m1 * r2[(l >= 1) ? l - 1 : 0], a20 = r2[l];
    float x0v = w00 * a0m2 + w01 * a0m1 + w02 * a00 + b0;
    float x1v = w10 * a1m2 + w11 * a1m1 + w12 * a10 + b1;
    float vr  = w20 * a2m2 + w21 * a2m1 + w22 * a20 + b2;
    x0loc[j] = x0v;
    vloc[j] = vr * x1v;
  }
  for (int i = tid; i < N2; i += 256) { re[i] = 0.f; im[i] = 0.f; }
  __syncthreads();
  for (int j = 0; j < 16; ++j) re[j * 256 + tid] = vloc[j];
  __syncthreads();
  fft_dif_fwd(re, im, tid);
  const float2* kd = kf + (size_t)d * N2;
  for (int i = tid; i < N2; i += 256) {     // both spectra are in bitrev order -> index-aligned
    float2 kw = kd[i];
    float xr = re[i], xi = im[i];
    re[i] = xr * kw.x - xi * kw.y;
    im[i] = xr * kw.y + xi * kw.x;
  }
  __syncthreads();
  fft_dit_inv(re, im, tid);
  bf16* o = yg + ((size_t)b * DIM + d) * L_SEQ;
  const float inv = 1.0f / (float)N2;
  for (int j = 0; j < 16; ++j) {
    int l = j * 256 + tid;
    float y = re[l] * inv + vloc[j] * cb;
    o[l] = (bf16)(y * x0loc[j]);
  }
}

// ---------------- GEMM2: out[l][b][e] = sum_d yg[b][d][l]*Wout[e,d] (A transposed on LDS stage) ----------------
__global__ __launch_bounds__(256) void gemm2_kernel(const bf16* __restrict__ Ayg,  // [B][D][L]
                                                    const bf16* __restrict__ Bw,   // wout [D][D]
                                                    float* __restrict__ out) {     // [L][B][D]
  const int b = blockIdx.z;
  const int mBase = blockIdx.x * 64;     // l
  const int nBase = blockIdx.y * 128;    // e
  const bf16* Ab = Ayg + (size_t)b * DIM * L_SEQ;

  __shared__ bf16 As[64 * 40];
  __shared__ bf16 Bs[128 * 40];

  const int tid = threadIdx.x;
  const int lane = tid & 31;
  const int wave = tid >> 5;
  const int wm = (wave >> 2) * 32;
  const int wn = (wave & 3) * 32;
  const int ln = lane & 15;
  const int hi = lane >> 4;

  v8f acc[2][2] = {};

  const int kk = tid >> 3;           // 0..31 (K row within tile)
  const int mm = (tid & 7) * 8;      // 0..56 (M chunk)

  for (int kt = 0; kt < DIM; kt += 32) {
    // A is [K][M]: coalesced load along m, transpose into As[m][k]
    bf16 tmp[8];
    *(uint4*)tmp = *(const uint4*)(Ab + (size_t)(kt + kk) * L_SEQ + mBase + mm);
    for (int i = 0; i < 8; ++i) As[(mm + i) * 40 + kk] = tmp[i];
#if defined(HAVE_ASYNC_LDS)
    for (int i = 0; i < 2; ++i) {
      int ch = tid + i * 256;
      int brr = ch >> 2, bcc = (ch & 3) * 8;
      __builtin_amdgcn_global_load_async_to_lds_b128(
          GPTR(Bw + (size_t)(nBase + brr) * DIM + kt + bcc), LPTR(&Bs[brr * 40 + bcc]), 0, 0);
    }
    async_wait0();
#else
    for (int i = 0; i < 2; ++i) {
      int ch = tid + i * 256;
      int brr = ch >> 2, bcc = (ch & 3) * 8;
      uint4 bv = *(const uint4*)(Bw + (size_t)(nBase + brr) * DIM + kt + bcc);
      *(uint4*)&Bs[brr * 40 + bcc] = bv;
    }
#endif
    if (kt + 32 < DIM) {
      __builtin_prefetch(Ab + (size_t)(kt + 32 + kk) * L_SEQ + mBase + mm, 0, 0);
      __builtin_prefetch(Bw + (size_t)(nBase + (tid >> 2)) * DIM + kt + 32, 0, 0);
    }
    __syncthreads();

    union FR { uint4 q[2]; v16bf v; } fa[2], fb[2];
    const int k0a = hi * 8;
    for (int s = 0; s < 2; ++s) {
      int row = wm + s * 16 + ln;
      fa[s].q[0] = *(const uint4*)&As[row * 40 + k0a];
      fa[s].q[1] = *(const uint4*)&As[row * 40 + k0a + 16];
    }
    const int k0b = hi * 8;
    for (int s = 0; s < 2; ++s) {
      int row = wn + s * 16 + ln;
      fb[s].q[0] = *(const uint4*)&Bs[row * 40 + k0b];
      fb[s].q[1] = *(const uint4*)&Bs[row * 40 + k0b + 16];
    }
    for (int si = 0; si < 2; ++si)
      for (int sj = 0; sj < 2; ++sj)
        acc[si][sj] = __builtin_amdgcn_wmma_f32_16x16x32_bf16(
            false, fa[si].v, false, fb[sj].v, (short)0, acc[si][sj], false, false);
    __syncthreads();
  }

  // out[l][b][e]; half-wave writes 16 consecutive e -> 64B contiguous per store group
  for (int si = 0; si < 2; ++si)
    for (int sj = 0; sj < 2; ++sj) {
      int n = nBase + wn + sj * 16 + ln;
      int m0 = mBase + wm + si * 16 + hi * 8;
      v8f c = acc[si][sj];
      for (int r = 0; r < 8; ++r)
        out[((size_t)(m0 + r) * BATCH + b) * DIM + n] = c[r];
    }
}

// ---------------- copy b_out to output tail ----------------
__global__ __launch_bounds__(256) void copy_bias_kernel(const float* __restrict__ bo,
                                                        float* __restrict__ dst) {
  int i = blockIdx.x * 256 + threadIdx.x;
  if (i < DIM) dst[i] = bo[i];
}

extern "C" void kernel_launch(void* const* d_in, const int* in_sizes, int n_in,
                              void* d_out, int out_size, void* d_ws, size_t ws_size,
                              hipStream_t stream) {
  const float* u    = (const float*)d_in[0];
  const float* gam  = (const float*)d_in[1];
  const float* bet  = (const float*)d_in[2];
  const float* Win  = (const float*)d_in[3];
  const float* bin  = (const float*)d_in[4];
  const float* sw   = (const float*)d_in[5];
  const float* sb   = (const float*)d_in[6];
  const float* kk   = (const float*)d_in[7];
  const float* cb   = (const float*)d_in[8];
  const float* Wout = (const float*)d_in[9];
  const float* bo   = (const float*)d_in[10];
  float* out = (float*)d_out;

  char* ws = (char*)d_ws;
  size_t off = 0;
  auto carve = [&](size_t bytes) -> void* {
    void* p = ws + off;
    off += (bytes + 255) & ~(size_t)255;
    return p;
  };
  bf16*   xn    = (bf16*)  carve((size_t)L_SEQ * BATCH * DIM * sizeof(bf16));
  bf16*   winb  = (bf16*)  carve((size_t)E3 * DIM * sizeof(bf16));
  bf16*   woutb = (bf16*)  carve((size_t)DIM * DIM * sizeof(bf16));
  float*  upT   = (float*) carve((size_t)BATCH * E3 * L_SEQ * sizeof(float));
  float2* kf    = (float2*)carve((size_t)DIM * N2 * sizeof(float2));
  bf16*   yg    = (bf16*)  carve((size_t)BATCH * DIM * L_SEQ * sizeof(bf16));

  cvt_bf16_kernel<<<4096, 256, 0, stream>>>(Win,  winb,  E3 * DIM);
  cvt_bf16_kernel<<<2048, 256, 0, stream>>>(Wout, woutb, DIM * DIM);
  ln_kernel<<<L_SEQ * BATCH, 256, 0, stream>>>(u, gam, bet, xn);
  gemm1_kernel<<<dim3(L_SEQ / 64, E3 / 128, BATCH), 256, 0, stream>>>(xn, winb, bin, upT);
  fftk_kernel<<<DIM, 256, 0, stream>>>(kk, kf);
  channel_kernel<<<dim3(DIM, BATCH), 256, 0, stream>>>(upT, sw, sb, kf, cb, yg);
  gemm2_kernel<<<dim3(L_SEQ / 64, DIM / 128, BATCH), 256, 0, stream>>>(yg, woutb, out);
  copy_bias_kernel<<<(DIM + 255) / 256, 256, 0, stream>>>(bo, out + (size_t)L_SEQ * BATCH * DIM);
}
